// Encoders_13451837571783
// MI455X (gfx1250) — matrix operand order repaired and checked by reading, not compiled, not run on hardware
//
#include <hip/hip_runtime.h>
#include <cstdint>
#include <cstddef>

// Problem constants (match reference)
#define Bsz 4
#define Ssz 1024
#define Dsz 512
#define Hsz 8
#define DEPTH 64
#define DFFsz 2048
#define NLAYERS 4

typedef __attribute__((ext_vector_type(16))) _Float16 v16h;
typedef __attribute__((ext_vector_type(8)))  float    v8f;
typedef int v4i_t __attribute__((vector_size(16)));   // matches builtin's V4i

union Frag16 { uint4 q[2]; v16h v; };

// ---------------------------------------------------------------------------
// Async copy of one 16-byte chunk (8 f16) global -> LDS, with sync fallback.
// (Confirmed lowering to global_load_async_to_lds_b128 on this toolchain.)
// ---------------------------------------------------------------------------
#if defined(__has_builtin)
#if __has_builtin(__builtin_amdgcn_global_load_async_to_lds_b128)
#define ASYNC_COPY 1
#endif
#endif
#ifndef ASYNC_COPY
#define ASYNC_COPY 0
#endif

__device__ __forceinline__ void copy16B_to_lds(_Float16* ldst,
                                               const _Float16* gsrc) {
#if ASYNC_COPY
  __builtin_amdgcn_global_load_async_to_lds_b128(
      (__attribute__((address_space(1))) v4i_t*)(void*)gsrc,
      (__attribute__((address_space(3))) v4i_t*)(void*)ldst,
      /*offset=*/0, /*cpol=*/0);
#else
  *reinterpret_cast<uint4*>(ldst) = *reinterpret_cast<const uint4*>(gsrc);
#endif
}

__device__ __forceinline__ void async_fence() {
#if ASYNC_COPY
#if defined(__has_builtin) && __has_builtin(__builtin_amdgcn_s_wait_asynccnt)
  __builtin_amdgcn_s_wait_asynccnt(0);
#else
  asm volatile("s_wait_asynccnt 0" ::: "memory");
#endif
#endif
}

// ---------------------------------------------------------------------------
// Stage a 64-row x 32-col f16 tile into LDS as [row][32]. src points at tile
// row 0; rows are `rstride` elements apart; k-window starts at kb.
// 128 threads, each moves two 16B chunks (rows tid>>2 and (tid>>2)+32).
// ---------------------------------------------------------------------------
__device__ __forceinline__ void stage_tile(const _Float16* __restrict__ src,
                                           size_t rstride, int kb,
                                           _Float16* lds) {
  const int row = threadIdx.x >> 2;
  const int seg = (threadIdx.x & 3) * 8;
  copy16B_to_lds(lds + row * 32 + seg, src + (size_t)row * rstride + kb + seg);
  copy16B_to_lds(lds + (row + 32) * 32 + seg,
                 src + (size_t)(row + 32) * rstride + kb + seg);
}

// ---------------------------------------------------------------------------
// WMMA core: 4 waves per 128-thread WG; wave w owns rows [16w,16w+16) of a
// 64x64 tile. As = [64 m][32 k], Bs = [64 n][32 k] f16 (B transposed layout).
// Per-lane fragments are contiguous uint4 pairs matching CDNA5 16-bit layout.
// ---------------------------------------------------------------------------
__device__ __forceinline__ void wmma_accum_64x64(const _Float16* As,
                                                 const _Float16* Bs,
                                                 v8f acc[4]) {
  const int lane = threadIdx.x & 31;
  const int wave = threadIdx.x >> 5;
  const int half = lane >> 4;
  const int l16  = lane & 15;

  Frag16 a;
  const _Float16* abase = As + (wave * 16 + l16) * 32 + half * 8;
  a.q[0] = *reinterpret_cast<const uint4*>(abase);
  a.q[1] = *reinterpret_cast<const uint4*>(abase + 16);

#pragma unroll
  for (int ns = 0; ns < 4; ++ns) {
    Frag16 b;
    const _Float16* bbase = Bs + (ns * 16 + l16) * 32 + half * 8;
    b.q[0] = reinterpret_cast<const uint4*>(bbase)[0];
    b.q[1] = *reinterpret_cast<const uint4*>(bbase + 16);
    acc[ns] = __builtin_amdgcn_wmma_f32_16x16x32_f16(
        false, a.v, false, b.v, (short)0, acc[ns], false, false);
  }
}

// Sign-split integer atomic max on float (works for mixed signs, init -inf).
__device__ __forceinline__ void atomicMaxFloat(float* addr, float val) {
  if (val >= 0.f)
    atomicMax((int*)addr, __float_as_int(val));
  else
    atomicMin((unsigned int*)addr, __float_as_uint(val));
}

// ---------------------------------------------------------------------------
// GEMM: C = act(A[M,K] @ W + bias), A f16 rows, WT f16 pre-transposed [N,K].
// K unrolled x2: two 32-deep slabs per barrier round (8 WMMAs per round).
// OUTMODE 0: C f16 [M,N].  OUTMODE 1 (V projection): C written head-
// transposed as VT[((b*H+h)*64+d)*S + i]  (contiguous along r).
// Grid: (N/64, M/64), block 128.
// ---------------------------------------------------------------------------
template <bool RELU, int OUTMODE>
__global__ __launch_bounds__(128) void gemm16_kernel(
    const _Float16* __restrict__ A, const _Float16* __restrict__ WT,
    const float* __restrict__ bias, _Float16* __restrict__ C, int N, int K) {
  __shared__ __align__(16) _Float16 As0[64 * 32];
  __shared__ __align__(16) _Float16 Bs0[64 * 32];
  __shared__ __align__(16) _Float16 As1[64 * 32];
  __shared__ __align__(16) _Float16 Bs1[64 * 32];
  const int n0 = blockIdx.x * 64;
  const int m0 = blockIdx.y * 64;

  v8f zero = {0.f, 0.f, 0.f, 0.f, 0.f, 0.f, 0.f, 0.f};
  v8f acc[4] = {zero, zero, zero, zero};

  const _Float16* Abase = A + (size_t)m0 * K;
  const _Float16* Bbase = WT + (size_t)n0 * K;
  for (int kb = 0; kb < K; kb += 64) {
    stage_tile(Abase, K, kb, As0);
    stage_tile(Bbase, K, kb, Bs0);
    stage_tile(Abase, K, kb + 32, As1);
    stage_tile(Bbase, K, kb + 32, Bs1);
    async_fence();
    __syncthreads();
    wmma_accum_64x64(As0, Bs0, acc);
    wmma_accum_64x64(As1, Bs1, acc);
    __syncthreads();
  }

  const int lane = threadIdx.x & 31, wave = threadIdx.x >> 5;
  const int half = lane >> 4, l16 = lane & 15;
  const int rowbase = m0 + wave * 16 + half * 8;
#pragma unroll
  for (int ns = 0; ns < 4; ++ns) {
    const int col = n0 + ns * 16 + l16;
    const float bv = bias[col];
#pragma unroll
    for (int r = 0; r < 8; ++r) {
      float v = acc[ns][r] + bv;
      if (RELU) v = fmaxf(v, 0.f);
      if (OUTMODE == 0) {
        C[(size_t)(rowbase + r) * N + col] = (_Float16)v;
      } else {
        const int t = rowbase + r;                 // global token
        const int b = t >> 10, i = t & (Ssz - 1);  // S = 1024
        const int h = col >> 6, d = col & 63;
        C[(size_t)((b * Hsz + h) * DEPTH + d) * Ssz + i] = (_Float16)v;
      }
    }
  }
}

// ---------------------------------------------------------------------------
// logits[b,h,i,j] = (Q_h[i]·K_h[j] + mask[b,i,j]*(-1e9)) * 0.125   (fp32 out)
// Also fuses the global-max reduction: per-block tile max -> atomic into
// mx[bh] (saves a full 128 MB logits read pass).
// Grid: (S/64 j, S/64 i, B*H), block 128.
// ---------------------------------------------------------------------------
__global__ __launch_bounds__(128) void attn_logits16_kernel(
    const _Float16* __restrict__ Q, const _Float16* __restrict__ Kt,
    const float* __restrict__ mask, float* __restrict__ out,
    float* __restrict__ mx) {
  __shared__ __align__(16) _Float16 As0[64 * 32];
  __shared__ __align__(16) _Float16 Bs0[64 * 32];
  __shared__ __align__(16) _Float16 As1[64 * 32];
  __shared__ __align__(16) _Float16 Bs1[64 * 32];
  __shared__ float redf[128];
  const int bh = blockIdx.z;
  const int b = bh / Hsz, h = bh % Hsz;
  const int j0 = blockIdx.x * 64;
  const int i0 = blockIdx.y * 64;
  const _Float16* Qb = Q + (size_t)(b * Ssz + i0) * Dsz + h * DEPTH;
  const _Float16* Kb = Kt + (size_t)(b * Ssz + j0) * Dsz + h * DEPTH;

  v8f zero = {0.f, 0.f, 0.f, 0.f, 0.f, 0.f, 0.f, 0.f};
  v8f acc[4] = {zero, zero, zero, zero};

  // DEPTH = 64: single unrolled round.
  stage_tile(Qb, Dsz, 0, As0);
  stage_tile(Kb, Dsz, 0, Bs0);
  stage_tile(Qb, Dsz, 32, As1);
  stage_tile(Kb, Dsz, 32, Bs1);
  async_fence();
  __syncthreads();
  wmma_accum_64x64(As0, Bs0, acc);
  wmma_accum_64x64(As1, Bs1, acc);

  const int tid = threadIdx.x;
  const int lane = tid & 31, wave = tid >> 5;
  const int half = lane >> 4, l16 = lane & 15;
  const int rowbase = i0 + wave * 16 + half * 8;
  float* ob = out + (size_t)bh * Ssz * Ssz;
  const float* mb = mask + (size_t)b * Ssz * Ssz;
  float lm = -3.4e38f;
#pragma unroll
  for (int ns = 0; ns < 4; ++ns) {
    const int j = j0 + ns * 16 + l16;
#pragma unroll
    for (int r = 0; r < 8; ++r) {
      const size_t idx = (size_t)(rowbase + r) * Ssz + j;
      const float v = (acc[ns][r] + mb[idx] * (-1e9f)) * 0.125f;  // mask, scale
      ob[idx] = v;
      lm = fmaxf(lm, v);
    }
  }
  // block max -> global atomic
  __syncthreads();
  redf[tid] = lm;
  __syncthreads();
  for (int k = 64; k > 0; k >>= 1) {
    if (tid < k) redf[tid] = fmaxf(redf[tid], redf[tid + k]);
    __syncthreads();
  }
  if (tid == 0) atomicMaxFloat(&mx[bh], redf[0]);
}

// ---------------------------------------------------------------------------
// ctx[b,j,h*64+d] = sum_i aw[b,h,i,j] * V[b,i,h*64+d]
// A rows = awT[b,h,j][i] f16, B rows = VT[b,h,d][i] f16 -> pure row copies.
// Grid: (1, S/64 j, B*H), block 128.
// ---------------------------------------------------------------------------
__global__ __launch_bounds__(128) void attn_av16_kernel(
    const _Float16* __restrict__ awT, const _Float16* __restrict__ VT,
    _Float16* __restrict__ ctx) {
  __shared__ __align__(16) _Float16 As0[64 * 32];
  __shared__ __align__(16) _Float16 Bs0[64 * 32];
  __shared__ __align__(16) _Float16 As1[64 * 32];
  __shared__ __align__(16) _Float16 Bs1[64 * 32];
  const int bh = blockIdx.z;
  const int b = bh / Hsz, h = bh % Hsz;
  const int j0 = blockIdx.y * 64;
  const _Float16* Ab = awT + (size_t)bh * Ssz * Ssz + (size_t)j0 * Ssz;
  const _Float16* Bb = VT + (size_t)bh * DEPTH * Ssz;

  v8f zero = {0.f, 0.f, 0.f, 0.f, 0.f, 0.f, 0.f, 0.f};
  v8f acc[4] = {zero, zero, zero, zero};

  for (int kb = 0; kb < Ssz; kb += 64) {
    stage_tile(Ab, Ssz, kb, As0);
    stage_tile(Bb, Ssz, kb, Bs0);
    stage_tile(Ab, Ssz, kb + 32, As1);
    stage_tile(Bb, Ssz, kb + 32, Bs1);
    async_fence();
    __syncthreads();
    wmma_accum_64x64(As0, Bs0, acc);
    wmma_accum_64x64(As1, Bs1, acc);
    __syncthreads();
  }

  const int lane = threadIdx.x & 31, wave = threadIdx.x >> 5;
  const int half = lane >> 4, l16 = lane & 15;
  const int rowbase = j0 + wave * 16 + half * 8;
#pragma unroll
  for (int ns = 0; ns < 4; ++ns) {
    const int d = ns * 16 + l16;
#pragma unroll
    for (int r = 0; r < 8; ++r) {
      ctx[(size_t)(b * Ssz + rowbase + r) * Dsz + h * DEPTH + d] =
          (_Float16)acc[ns][r];
    }
  }
}

// ---------------------------------------------------------------------------
// Softmax: sum of exp over the full SxS map per (b,h) (max already fused).
// ---------------------------------------------------------------------------
__global__ __launch_bounds__(256) void softmax_sum_kernel(
    const float* __restrict__ lg, const float* __restrict__ mx,
    float* __restrict__ sm) {
  const int bh = blockIdx.x, tid = threadIdx.x;
  const float m = mx[bh];
  const float4* p = reinterpret_cast<const float4*>(lg + (size_t)bh * Ssz * Ssz);
  float s = 0.f;
  for (int i = tid; i < Ssz * Ssz / 4; i += 256) {
    float4 v = p[i];
    s += __expf(v.x - m) + __expf(v.y - m) + __expf(v.z - m) + __expf(v.w - m);
  }
  __shared__ float red[256];
  red[tid] = s; __syncthreads();
  for (int st = 128; st > 0; st >>= 1) {
    if (tid < st) red[tid] += red[tid + st];
    __syncthreads();
  }
  if (tid == 0) sm[bh] = red[0];
}

// Finalize: aw = exp(l-m)/sum * row_count (fp32, in place) AND emit a
// transposed f16 copy awT[b,h,j,i] via an LDS tile (all accesses coalesced).
// Grid: (S/64 j, S/64 i, B*H), block 256.
__global__ __launch_bounds__(256) void softmax_finalize_T_kernel(
    float* __restrict__ aw, _Float16* __restrict__ awT,
    const float* __restrict__ mx, const float* __restrict__ sm,
    const float* __restrict__ rc) {
  __shared__ _Float16 t[64][65];
  const int bh = blockIdx.z, tid = threadIdx.x;
  const int j0 = blockIdx.x * 64, i0 = blockIdx.y * 64;
  const float m = mx[bh];
  const float scale = rc[0] / sm[bh];
  float* awb = aw + (size_t)bh * Ssz * Ssz;
  for (int e = tid; e < 64 * 64; e += 256) {
    const int ii = e >> 6, jj = e & 63;
    const size_t idx = (size_t)(i0 + ii) * Ssz + j0 + jj;
    const float v = __expf(awb[idx] - m) * scale;
    awb[idx] = v;
    t[jj][ii] = (_Float16)v;
  }
  __syncthreads();
  _Float16* awTb = awT + (size_t)bh * Ssz * Ssz;
  for (int e = tid; e < 64 * 64; e += 256) {
    const int jj = e >> 6, ii = e & 63;
    awTb[(size_t)(j0 + jj) * Ssz + i0 + ii] = t[jj][ii];
  }
}

// ---------------------------------------------------------------------------
// out = LayerNorm(a + b) * g + beta (row of D=512), f16 in, f16/f32 out.
// ---------------------------------------------------------------------------
template <typename OutT>
__global__ __launch_bounds__(128) void add_ln16_kernel(
    const _Float16* __restrict__ a, const _Float16* __restrict__ bsrc,
    const float* __restrict__ g, const float* __restrict__ be,
    OutT* __restrict__ out) {
  const int row = blockIdx.x, tid = threadIdx.x;
  const _Float16* pa = a + (size_t)row * Dsz;
  const _Float16* pb = bsrc + (size_t)row * Dsz;
  float v[4];
  float s = 0.f;
#pragma unroll
  for (int i = 0; i < 4; ++i) {
    const int c = tid + 128 * i;
    v[i] = (float)pa[c] + (float)pb[c];
    s += v[i];
  }
  __shared__ float red[128];
  red[tid] = s; __syncthreads();
  for (int k = 64; k > 0; k >>= 1) {
    if (tid < k) red[tid] += red[tid + k];
    __syncthreads();
  }
  const float mean = red[0] / (float)Dsz;
  __syncthreads();
  float sv = 0.f;
#pragma unroll
  for (int i = 0; i < 4; ++i) { const float d = v[i] - mean; sv += d * d; }
  red[tid] = sv; __syncthreads();
  for (int k = 64; k > 0; k >>= 1) {
    if (tid < k) red[tid] += red[tid + k];
    __syncthreads();
  }
  const float inv = rsqrtf(red[0] / (float)Dsz + 1e-9f);
#pragma unroll
  for (int i = 0; i < 4; ++i) {
    const int c = tid + 128 * i;
    out[(size_t)row * Dsz + c] = (OutT)(g[c] * (v[i] - mean) * inv + be[c]);
  }
}

// ---------------------------------------------------------------------------
// Pre-pass: fp32 -> f16 elementwise (x), and transposed f16 weights.
// ---------------------------------------------------------------------------
__global__ __launch_bounds__(256) void cvt16_kernel(
    const float* __restrict__ in, _Float16* __restrict__ out) {
  const size_t g = (size_t)blockIdx.x * blockDim.x + threadIdx.x;  // float4 idx
  const float4 v = reinterpret_cast<const float4*>(in)[g];
  union { _Float16 h[4]; uint2 u; } p;
  p.h[0] = (_Float16)v.x; p.h[1] = (_Float16)v.y;
  p.h[2] = (_Float16)v.z; p.h[3] = (_Float16)v.w;
  reinterpret_cast<uint2*>(out)[g] = p.u;
}

// W[K,N] fp32 -> WT[N,K] f16. Grid (N/64, K/64), block 256.
__global__ __launch_bounds__(256) void wtransp_kernel(
    const float* __restrict__ W, _Float16* __restrict__ WT, int K, int N) {
  __shared__ _Float16 t[64][65];
  const int tid = threadIdx.x;
  const int n0 = blockIdx.x * 64, k0 = blockIdx.y * 64;
  for (int e = tid; e < 64 * 64; e += 256) {
    const int kk = e >> 6, nn = e & 63;
    t[kk][nn] = (_Float16)W[(size_t)(k0 + kk) * N + n0 + nn];
  }
  __syncthreads();
  for (int e = tid; e < 64 * 64; e += 256) {
    const int nn = e >> 6, kk = e & 63;
    WT[(size_t)(n0 + nn) * K + k0 + kk] = t[kk][nn];
  }
}

__global__ __launch_bounds__(256) void rowcount_kernel(
    const int* __restrict__ protok, float* __restrict__ out) {
  const int tid = threadIdx.x;
  int c = 0;
  for (int i = tid; i < Ssz; i += 256) c += (protok[i] != 0) ? 1 : 0;
  __shared__ int red[256];
  red[tid] = c; __syncthreads();
  for (int s = 128; s > 0; s >>= 1) {
    if (tid < s) red[tid] += red[tid + s];
    __syncthreads();
  }
  if (tid == 0) out[0] = (float)red[0];
}

// Per-layer reset of softmax statistics (mx = -inf, sm = 0).
__global__ void init_stats_kernel(float* __restrict__ mx,
                                  float* __restrict__ sm) {
  const int i = threadIdx.x;
  if (i < Bsz * Hsz) {
    mx[i] = -3.4e38f;
    sm[i] = 0.f;
  }
}

// ---------------------------------------------------------------------------
extern "C" void kernel_launch(void* const* d_in, const int* in_sizes, int n_in,
                              void* d_out, int out_size, void* d_ws,
                              size_t ws_size, hipStream_t stream) {
  (void)in_sizes; (void)n_in; (void)out_size; (void)ws_size;

  const float* x_in   = (const float*)d_in[0];
  const float* maskp  = (const float*)d_in[1];
  const int*   protok = (const int*)d_in[2];
  const float* wq_w = (const float*)d_in[3];
  const float* wq_b = (const float*)d_in[4];
  const float* wk_w = (const float*)d_in[5];
  const float* wk_b = (const float*)d_in[6];
  const float* wv_w = (const float*)d_in[7];
  const float* wv_b = (const float*)d_in[8];
  const float* wo_w = (const float*)d_in[9];
  const float* wo_b = (const float*)d_in[10];
  const float* f_w1 = (const float*)d_in[11];
  const float* f_b1 = (const float*)d_in[12];
  const float* f_w2 = (const float*)d_in[13];
  const float* f_b2 = (const float*)d_in[14];
  const float* ln1g = (const float*)d_in[15];
  const float* ln1b = (const float*)d_in[16];
  const float* ln2g = (const float*)d_in[17];
  const float* ln2b = (const float*)d_in[18];

  const size_t NTOK = (size_t)Bsz * Ssz;          // 4096
  const size_t MAT  = NTOK * Dsz;                 // 2,097,152
  const size_t AWSZ = (size_t)Bsz * Hsz * Ssz * Ssz;  // 33,554,432

  float* out_x = (float*)d_out;                   // [B,S,D] fp32
  float* AW    = out_x + MAT;                     // [B,H,S,S] fp32 (output)

  _Float16* X16  = (_Float16*)d_ws;
  _Float16* Q16  = X16 + MAT;
  _Float16* K16  = Q16 + MAT;
  _Float16* VT16 = K16 + MAT;                     // [B,H,64,S]
  _Float16* O116 = VT16 + MAT;
  _Float16* HID  = O116 + MAT;                    // [4096,2048]
  _Float16* AWT  = HID + NTOK * DFFsz;            // [B,H,S,S] f16 transposed
  _Float16* WQT  = AWT + AWSZ;                    // [512,512]
  _Float16* WKT  = WQT + (size_t)Dsz * Dsz;
  _Float16* WVT  = WKT + (size_t)Dsz * Dsz;
  _Float16* WOT  = WVT + (size_t)Dsz * Dsz;
  _Float16* F1T  = WOT + (size_t)Dsz * Dsz;       // [2048,512]
  _Float16* F2T  = F1T + (size_t)Dsz * DFFsz;     // [512,2048]
  float*    RED  = (float*)(F2T + (size_t)Dsz * DFFsz);
  float* MX = RED + 1;
  float* SM = RED + 33;
  _Float16* CTX = Q16;                            // Q dead after logits
  _Float16* ATT = K16;                            // K dead after logits
  _Float16* FFN = K16;                            // attn_out dead after LN1

  // ---- pre-pass: conversions, weight transposes, row count ----
  cvt16_kernel<<<(int)(MAT / 1024), 256, 0, stream>>>(x_in, X16);
  {
    dim3 gDD(Dsz / 64, Dsz / 64);
    wtransp_kernel<<<gDD, 256, 0, stream>>>(wq_w, WQT, Dsz, Dsz);
    wtransp_kernel<<<gDD, 256, 0, stream>>>(wk_w, WKT, Dsz, Dsz);
    wtransp_kernel<<<gDD, 256, 0, stream>>>(wv_w, WVT, Dsz, Dsz);
    wtransp_kernel<<<gDD, 256, 0, stream>>>(wo_w, WOT, Dsz, Dsz);
    wtransp_kernel<<<dim3(DFFsz / 64, Dsz / 64), 256, 0, stream>>>(f_w1, F1T,
                                                                   Dsz, DFFsz);
    wtransp_kernel<<<dim3(Dsz / 64, DFFsz / 64), 256, 0, stream>>>(f_w2, F2T,
                                                                   DFFsz, Dsz);
  }
  rowcount_kernel<<<1, 256, 0, stream>>>(protok, RED);

  const dim3 gProj(Dsz / 64, (int)(NTOK / 64));      // (8, 64)
  const dim3 gFfn1(DFFsz / 64, (int)(NTOK / 64));    // (32, 64)
  const dim3 gAttn(Ssz / 64, Ssz / 64, Bsz * Hsz);   // (16, 16, 32)
  const dim3 gAV(1, Ssz / 64, Bsz * Hsz);            // (1, 16, 32)

  for (int layer = 0; layer < NLAYERS; ++layer) {
    init_stats_kernel<<<1, 64, 0, stream>>>(MX, SM);

    gemm16_kernel<false, 0><<<gProj, 128, 0, stream>>>(X16, WQT, wq_b, Q16,
                                                       Dsz, Dsz);
    gemm16_kernel<false, 0><<<gProj, 128, 0, stream>>>(X16, WKT, wk_b, K16,
                                                       Dsz, Dsz);
    gemm16_kernel<false, 1><<<gProj, 128, 0, stream>>>(X16, WVT, wv_b, VT16,
                                                       Dsz, Dsz);

    attn_logits16_kernel<<<gAttn, 128, 0, stream>>>(Q16, K16, maskp, AW, MX);
    softmax_sum_kernel<<<Bsz * Hsz, 256, 0, stream>>>(AW, MX, SM);
    softmax_finalize_T_kernel<<<gAttn, 256, 0, stream>>>(AW, AWT, MX, SM, RED);

    attn_av16_kernel<<<gAV, 128, 0, stream>>>(AWT, VT16, CTX);

    gemm16_kernel<false, 0><<<gProj, 128, 0, stream>>>(CTX, WOT, wo_b, ATT,
                                                       Dsz, Dsz);
    add_ln16_kernel<_Float16><<<(int)NTOK, 128, 0, stream>>>(X16, ATT, ln1g,
                                                             ln1b, O116);

    gemm16_kernel<true, 0><<<gFfn1, 128, 0, stream>>>(O116, F1T, f_b1, HID,
                                                      DFFsz, Dsz);
    gemm16_kernel<false, 0><<<gProj, 128, 0, stream>>>(HID, F2T, f_b2, FFN,
                                                       Dsz, DFFsz);

    if (layer == NLAYERS - 1) {
      add_ln16_kernel<float><<<(int)NTOK, 128, 0, stream>>>(O116, FFN, ln2g,
                                                            ln2b, out_x);
    } else {
      add_ln16_kernel<_Float16><<<(int)NTOK, 128, 0, stream>>>(O116, FFN, ln2g,
                                                               ln2b, X16);
    }
  }
}
